// SelfAttention_103079215673
// MI455X (gfx1250) — matrix operand order repaired and checked by reading, not compiled, not run on hardware
//
#include <hip/hip_runtime.h>
#include <hip/hip_bf16.h>

#define DEVINL __device__ __forceinline__

constexpr int Hdim  = 1024;
constexpr int NH    = 16;
constexpr int HD    = 64;     // head dim
constexpr int Bsz   = 4;
constexpr int Sdim  = 2048;
constexpr int MROWS = Bsz * Sdim;   // 8192

typedef __attribute__((ext_vector_type(16))) __bf16 v16bf;
typedef __attribute__((ext_vector_type(8)))  float  v8f;

union Frag { unsigned int u[8]; v16bf v; };

DEVINL unsigned short f2bf(float f) {
  unsigned int u = __float_as_uint(f);
  u += 0x7FFFu + ((u >> 16) & 1u);          // round-to-nearest-even
  return (unsigned short)(u >> 16);
}
// packed f32x2 -> bf16x2 (lowers to v_cvt_pk_bf16_f32)
DEVINL unsigned int pack2(float lo, float hi) {
  __hip_bfloat162 h2 = __float22bfloat162_rn(make_float2(lo, hi));
  unsigned int u;
  __builtin_memcpy(&u, &h2, sizeof(u));
  return u;
}

DEVINL v8f wmma_bf16(const Frag& a, const Frag& b, v8f c) {
  return __builtin_amdgcn_wmma_f32_16x16x32_bf16(
      /*neg_a=*/false, a.v, /*neg_b=*/false, b.v,
      /*c_mod=*/(short)0, c, /*reuse_a=*/false, /*reuse_b=*/false);
}

// 16B LDS fragment slice load
DEVINL void ld4(Frag& f, int base, const unsigned int* p) {
  uint4 x = *reinterpret_cast<const uint4*>(p);
  f.u[base + 0] = x.x; f.u[base + 1] = x.y; f.u[base + 2] = x.z; f.u[base + 3] = x.w;
}

// CDNA5 async global->LDS copy (ASYNCcnt-tracked, no VGPR data path)
DEVINL void async_copy16(unsigned int lds_byte_off, const void* gptr) {
  asm volatile("global_load_async_to_lds_b128 %0, %1, off"
               :: "v"(lds_byte_off), "v"((unsigned long long)(uintptr_t)gptr)
               : "memory");
}
DEVINL void async_wait0() {
  asm volatile("s_wait_asynccnt 0" ::: "memory");
}

// ---------------------------------------------------------------------------
// Kernel 1: fused QKV projection.  C[8192,1024] = in @ W + b  (gridDim.z picks
// Wq/Wk/Wv).  Output scattered to [B,NH,S,HD] bf16.  128x128 tile, 8 waves.
// ---------------------------------------------------------------------------
__global__ __launch_bounds__(256) void qkv_gemm_kernel(
    const float* __restrict__ in,
    const float* __restrict__ Wq, const float* __restrict__ bq,
    const float* __restrict__ Wk, const float* __restrict__ bk,
    const float* __restrict__ Wv, const float* __restrict__ bv,
    unsigned short* __restrict__ qo, unsigned short* __restrict__ ko,
    unsigned short* __restrict__ vo)
{
  __shared__ unsigned int aU[128][20];   // A tile [m][k-pair]  (16B-aligned rows)
  __shared__ unsigned int bU[128][20];   // B tile transposed [n][k-pair]

  const int n0 = blockIdx.x * 128;
  const int m0 = blockIdx.y * 128;
  const int which = blockIdx.z;
  const float* W    = (which == 0) ? Wq : (which == 1) ? Wk : Wv;
  const float* bias = (which == 0) ? bq : (which == 1) ? bk : bv;
  unsigned short* out = (which == 0) ? qo : (which == 1) ? ko : vo;

  const int tid = threadIdx.x;
  const int lane = tid & 31, wave = tid >> 5;
  const int hi = lane >> 4, ln = lane & 15;

  v8f acc[8] = {};

  const int arow = tid >> 3;           // 0..31
  const int acol = (tid & 7) * 4;      // 0..28

  for (int kt = 0; kt < Hdim / 32; ++kt) {
    // A tile 128x32 f32 -> packed bf16 pairs in LDS
    #pragma unroll
    for (int r = 0; r < 4; ++r) {
      int row = r * 32 + arow;
      const float4 f = *reinterpret_cast<const float4*>(
          in + (size_t)(m0 + row) * Hdim + kt * 32 + acol);
      *reinterpret_cast<uint2*>(&aU[row][acol >> 1]) =
          make_uint2(pack2(f.x, f.y), pack2(f.z, f.w));
    }
    // B tile 32x128 f32 -> transposed bf16 pairs: float4-pair scheme.
    // idx enumerates (k-pair, n-quad); 2 idx per thread.
    #pragma unroll
    for (int e = 0; e < 2; ++e) {
      int idx = tid * 2 + e;
      int kp = idx >> 5;               // 0..15 key-pair
      int n4 = (idx & 31) * 4;         // 0..124
      const float* base = W + (size_t)(kt * 32 + 2 * kp) * Hdim + n0 + n4;
      const float4 w0 = *reinterpret_cast<const float4*>(base);
      const float4 w1 = *reinterpret_cast<const float4*>(base + Hdim);
      const float* w0p = &w0.x; const float* w1p = &w1.x;
      #pragma unroll
      for (int j = 0; j < 4; ++j) bU[n4 + j][kp] = pack2(w0p[j], w1p[j]);
    }
    __syncthreads();

    Frag af;
    {
      int row = wave * 16 + ln;
      int pb = hi * 4;                       // half-wave K split
      ld4(af, 0, &aU[row][pb]);
      ld4(af, 4, &aU[row][pb + 8]);
    }
    #pragma unroll
    for (int nt = 0; nt < 8; ++nt) {
      Frag bf;
      int n = nt * 16 + ln;
      int pb = hi * 4;
      ld4(bf, 0, &bU[n][pb]);
      ld4(bf, 4, &bU[n][pb + 8]);
      acc[nt] = wmma_bf16(af, bf, acc[nt]);
    }
    __syncthreads();
  }

  // Epilogue: +bias, scatter to [B,NH,S,HD] bf16
  #pragma unroll
  for (int nt = 0; nt < 8; ++nt) {
    int n = n0 + nt * 16 + ln;
    float bv_ = bias[n];
    int h = n >> 6, d = n & 63;
    #pragma unroll
    for (int r = 0; r < 8; ++r) {
      int m = m0 + wave * 16 + hi * 8 + r;
      int b = m >> 11, s = m & (Sdim - 1);
      out[((size_t)((b * NH + h) * Sdim + s)) * HD + d] = f2bf(acc[nt][r] + bv_);
    }
  }
}

// ---------------------------------------------------------------------------
// Kernel 2: flash attention.  One workgroup = one head x 128 queries.
// Q/K tiles staged with global_load_async_to_lds_b128.
// ---------------------------------------------------------------------------
__global__ __launch_bounds__(256) void attention_kernel(
    const unsigned short* __restrict__ q,   // [B,NH,S,HD] bf16
    const unsigned short* __restrict__ k,
    const unsigned short* __restrict__ v,
    const float* __restrict__ mask,         // [B,1,1,S]
    unsigned short* __restrict__ ctx)       // [B,S,H] bf16
{
  __shared__ unsigned int qU[128][36];      // Q tile  [q][d-pair]
  __shared__ unsigned int kU[64][36];       // K tile  [key][d-pair]
  __shared__ unsigned int vtU[64][36];      // V^T     [d][key-pair]
  __shared__ unsigned int pU[8][16][36];    // per-wave P scratch [q][key-pair]

  const int q0 = blockIdx.x * 128;
  const int bh = blockIdx.y;
  const int b = bh / NH, h = bh % NH;
  const int tid = threadIdx.x, lane = tid & 31, wave = tid >> 5;
  const int hi = lane >> 4, ln = lane & 15;

  const unsigned int* qhead = reinterpret_cast<const unsigned int*>(q) + (size_t)bh * Sdim * (HD / 2);
  const unsigned int* khead = reinterpret_cast<const unsigned int*>(k) + (size_t)bh * Sdim * (HD / 2);
  const unsigned short* vhead = v + (size_t)bh * Sdim * HD;

  // Q tile: async copy 128 rows x 128B
  {
    int row = tid >> 1;
    int uo = (tid & 1) * 16;
    const unsigned int* src = qhead + (size_t)(q0 + row) * (HD / 2) + uo;
    #pragma unroll
    for (int j = 0; j < 4; ++j)
      async_copy16((unsigned int)(uintptr_t)&qU[row][uo + 4 * j], src + 4 * j);
  }

  float m_run[8], l_run[8];
  #pragma unroll
  for (int r = 0; r < 8; ++r) { m_run[r] = -1e30f; l_run[r] = 0.f; }
  v8f o[4] = {};

  const float scl = 0.125f;                      // 1/sqrt(64)
  const float LOG2E = 1.44269504088896340736f;

  for (int kblk = 0; kblk < Sdim / 64; ++kblk) {
    const int k0 = kblk * 64;
    __syncthreads();                              // prev readers done
    {   // K tile: async copy 64 rows x 128B
      int row = tid >> 2, uo = (tid & 3) * 8;
      const unsigned int* src = khead + (size_t)(k0 + row) * (HD / 2) + uo;
      async_copy16((unsigned int)(uintptr_t)&kU[row][uo],     src);
      async_copy16((unsigned int)(uintptr_t)&kU[row][uo + 4], src + 4);
    }
    {   // V^T tile: b128 loads + register repack -> vtU[d][key-pair]
      int kp = tid >> 3;            // 0..31
      int d8 = (tid & 7) * 8;       // 0..56
      const uint4 va = *reinterpret_cast<const uint4*>(vhead + (size_t)(k0 + 2 * kp    ) * HD + d8);
      const uint4 vb = *reinterpret_cast<const uint4*>(vhead + (size_t)(k0 + 2 * kp + 1) * HD + d8);
      const unsigned int* au = &va.x;
      const unsigned int* bu = &vb.x;
      #pragma unroll
      for (int j = 0; j < 8; ++j) {
        unsigned int lo  = (au[j >> 1] >> ((j & 1) * 16)) & 0xffffu;
        unsigned int hi2 = (bu[j >> 1] >> ((j & 1) * 16)) & 0xffffu;
        vtU[d8 + j][kp] = lo | (hi2 << 16);
      }
    }
    async_wait0();                 // ASYNCcnt drained before barrier
    __syncthreads();

    // ---- scores = Q K^T : 4 tiles of 16x16, K-dim = 64 (2 wmma steps) ----
    v8f sc[4] = {};
    #pragma unroll
    for (int kk = 0; kk < 2; ++kk) {
      Frag af;
      int row = wave * 16 + ln;
      int pb = kk * 16 + hi * 4;
      ld4(af, 0, &qU[row][pb]);
      ld4(af, 4, &qU[row][pb + 8]);
      #pragma unroll
      for (int t = 0; t < 4; ++t) {
        Frag bf;
        int n = t * 16 + ln;
        int pb2 = kk * 16 + hi * 4;
        ld4(bf, 0, &kU[n][pb2]);
        ld4(bf, 4, &kU[n][pb2 + 8]);
        sc[t] = wmma_bf16(af, bf, sc[t]);
      }
    }

    // ---- scale + mask + online softmax ----
    float mv[4];
    #pragma unroll
    for (int t = 0; t < 4; ++t) mv[t] = mask[(size_t)b * Sdim + k0 + t * 16 + ln];
    #pragma unroll
    for (int t = 0; t < 4; ++t)
      #pragma unroll
      for (int r = 0; r < 8; ++r) sc[t][r] = sc[t][r] * scl + mv[t];

    unsigned short* prow = reinterpret_cast<unsigned short*>(&pU[wave][0][0]);
    #pragma unroll
    for (int r = 0; r < 8; ++r) {
      float x = fmaxf(fmaxf(sc[0][r], sc[1][r]), fmaxf(sc[2][r], sc[3][r]));
      #pragma unroll
      for (int off = 8; off >= 1; off >>= 1) x = fmaxf(x, __shfl_xor(x, off, 32));
      float mn = fmaxf(m_run[r], x);
      float alpha = exp2f((m_run[r] - mn) * LOG2E);
      float p[4], ssum = 0.f;
      #pragma unroll
      for (int t = 0; t < 4; ++t) { p[t] = exp2f((sc[t][r] - mn) * LOG2E); ssum += p[t]; }
      #pragma unroll
      for (int off = 8; off >= 1; off >>= 1) ssum += __shfl_xor(ssum, off, 32);
      l_run[r] = l_run[r] * alpha + ssum;
      m_run[r] = mn;
      #pragma unroll
      for (int dt = 0; dt < 4; ++dt) o[dt][r] *= alpha;
      int m = hi * 8 + r;
      #pragma unroll
      for (int t = 0; t < 4; ++t) prow[m * 72 + t * 16 + ln] = f2bf(p[t]);  // C->A layout
    }

    // ---- O += P @ V : 2 wmma steps over 64 keys, 4 d-tiles ----
    #pragma unroll
    for (int kk = 0; kk < 2; ++kk) {
      Frag af;
      int pb = kk * 16 + hi * 4;
      ld4(af, 0, &pU[wave][ln][pb]);
      ld4(af, 4, &pU[wave][ln][pb + 8]);
      #pragma unroll
      for (int dt = 0; dt < 4; ++dt) {
        Frag bf;
        int d = dt * 16 + ln;
        int pb2 = kk * 16 + hi * 4;
        ld4(bf, 0, &vtU[d][pb2]);
        ld4(bf, 4, &vtU[d][pb2 + 8]);
        o[dt] = wmma_bf16(af, bf, o[dt]);
      }
    }
  }

  // normalize + store ctx as [B,S,H] bf16
  #pragma unroll
  for (int dt = 0; dt < 4; ++dt)
    #pragma unroll
    for (int r = 0; r < 8; ++r) {
      int m = hi * 8 + r;
      int qrow = q0 + wave * 16 + m;
      float val = o[dt][r] / l_run[r];
      ctx[((size_t)(b * Sdim + qrow)) * Hdim + h * HD + dt * 16 + ln] = f2bf(val);
    }
}

// ---------------------------------------------------------------------------
// Kernel 3: output projection.  d_out = ctx(bf16) @ Wd + bd  (fp32 out)
// A tile staged with async copies (pure bf16 memcpy).
// ---------------------------------------------------------------------------
__global__ __launch_bounds__(256) void proj_gemm_kernel(
    const unsigned short* __restrict__ ctx,
    const float* __restrict__ Wd, const float* __restrict__ bd,
    float* __restrict__ outp)
{
  __shared__ unsigned int aU[128][20];
  __shared__ unsigned int bU[128][20];

  const int n0 = blockIdx.x * 128;
  const int m0 = blockIdx.y * 128;
  const int tid = threadIdx.x, lane = tid & 31, wave = tid >> 5;
  const int hi = lane >> 4, ln = lane & 15;
  const unsigned int* ctxu = reinterpret_cast<const unsigned int*>(ctx);

  v8f acc[8] = {};
  const int arow = tid >> 1;
  const int auo  = (tid & 1) * 8;

  for (int kt = 0; kt < Hdim / 32; ++kt) {
    {   // A tile: async bf16 copy, 128 rows x 64B
      const unsigned int* src = ctxu + (size_t)(m0 + arow) * (Hdim / 2) + kt * 16 + auo;
      async_copy16((unsigned int)(uintptr_t)&aU[arow][auo],     src);
      async_copy16((unsigned int)(uintptr_t)&aU[arow][auo + 4], src + 4);
    }
    #pragma unroll
    for (int e = 0; e < 2; ++e) {
      int idx = tid * 2 + e;
      int kp = idx >> 5;
      int n4 = (idx & 31) * 4;
      const float* base = Wd + (size_t)(kt * 32 + 2 * kp) * Hdim + n0 + n4;
      const float4 w0 = *reinterpret_cast<const float4*>(base);
      const float4 w1 = *reinterpret_cast<const float4*>(base + Hdim);
      const float* w0p = &w0.x; const float* w1p = &w1.x;
      #pragma unroll
      for (int j = 0; j < 4; ++j) bU[n4 + j][kp] = pack2(w0p[j], w1p[j]);
    }
    async_wait0();
    __syncthreads();

    Frag af;
    {
      int row = wave * 16 + ln;
      int pb = hi * 4;
      ld4(af, 0, &aU[row][pb]);
      ld4(af, 4, &aU[row][pb + 8]);
    }
    #pragma unroll
    for (int nt = 0; nt < 8; ++nt) {
      Frag bf;
      int n = nt * 16 + ln;
      int pb = hi * 4;
      ld4(bf, 0, &bU[n][pb]);
      ld4(bf, 4, &bU[n][pb + 8]);
      acc[nt] = wmma_bf16(af, bf, acc[nt]);
    }
    __syncthreads();
  }

  #pragma unroll
  for (int nt = 0; nt < 8; ++nt) {
    int n = n0 + nt * 16 + ln;
    float bb = bd[n];
    #pragma unroll
    for (int r = 0; r < 8; ++r) {
      int m = m0 + wave * 16 + hi * 8 + r;
      outp[(size_t)m * Hdim + n] = acc[nt][r] + bb;
    }
  }
}

// ---------------------------------------------------------------------------
// Kernel 4: y = LayerNorm(y + x) in place on d_out.  One block per row.
// ---------------------------------------------------------------------------
__global__ __launch_bounds__(256) void add_ln_kernel(
    float* __restrict__ y, const float* __restrict__ x_in,
    const float* __restrict__ w, const float* __restrict__ bias)
{
  __shared__ float red[8];
  const int row = blockIdx.x;
  const int tid = threadIdx.x, lane = tid & 31, wave = tid >> 5;

  float xv[4], s = 0.f;
  #pragma unroll
  for (int j = 0; j < 4; ++j) {
    int c = tid + j * 256;
    xv[j] = y[(size_t)row * Hdim + c] + x_in[(size_t)row * Hdim + c];
    s += xv[j];
  }
  #pragma unroll
  for (int off = 16; off >= 1; off >>= 1) s += __shfl_xor(s, off, 32);
  if (lane == 0) red[wave] = s;
  __syncthreads();
  float t = (lane < 8) ? red[lane] : 0.f;
  #pragma unroll
  for (int off = 4; off >= 1; off >>= 1) t += __shfl_xor(t, off, 32);
  float mean = __shfl(t, 0, 32) / (float)Hdim;
  __syncthreads();

  float vs = 0.f;
  #pragma unroll
  for (int j = 0; j < 4; ++j) { float d = xv[j] - mean; vs += d * d; }
  #pragma unroll
  for (int off = 16; off >= 1; off >>= 1) vs += __shfl_xor(vs, off, 32);
  if (lane == 0) red[wave] = vs;
  __syncthreads();
  float t2 = (lane < 8) ? red[lane] : 0.f;
  #pragma unroll
  for (int off = 4; off >= 1; off >>= 1) t2 += __shfl_xor(t2, off, 32);
  float rstd = rsqrtf(__shfl(t2, 0, 32) / (float)Hdim + 1e-12f);

  #pragma unroll
  for (int j = 0; j < 4; ++j) {
    int c = tid + j * 256;
    y[(size_t)row * Hdim + c] = w[c] * ((xv[j] - mean) * rstd) + bias[c];
  }
}

// ---------------------------------------------------------------------------
extern "C" void kernel_launch(void* const* d_in, const int* in_sizes, int n_in,
                              void* d_out, int out_size, void* d_ws, size_t ws_size,
                              hipStream_t stream) {
  const float* input = (const float*)d_in[0];
  const float* mask  = (const float*)d_in[1];
  const float* Wq = (const float*)d_in[2];  const float* bq = (const float*)d_in[3];
  const float* Wk = (const float*)d_in[4];  const float* bk = (const float*)d_in[5];
  const float* Wv = (const float*)d_in[6];  const float* bv = (const float*)d_in[7];
  const float* Wd = (const float*)d_in[8];  const float* bd = (const float*)d_in[9];
  const float* lw = (const float*)d_in[10]; const float* lb = (const float*)d_in[11];
  float* out = (float*)d_out;

  char* ws = (char*)d_ws;
  const size_t BUF = (size_t)MROWS * Hdim * sizeof(unsigned short);  // 16.78 MB
  unsigned short* qb  = (unsigned short*)(ws);
  unsigned short* kb  = (unsigned short*)(ws + BUF);
  unsigned short* vb  = (unsigned short*)(ws + 2 * BUF);
  unsigned short* ctx = (unsigned short*)(ws + 3 * BUF);

  qkv_gemm_kernel<<<dim3(Hdim / 128, MROWS / 128, 3), 256, 0, stream>>>(
      input, Wq, bq, Wk, bk, Wv, bv, qb, kb, vb);
  attention_kernel<<<dim3(Sdim / 128, Bsz * NH), 256, 0, stream>>>(
      qb, kb, vb, mask, ctx);
  proj_gemm_kernel<<<dim3(Hdim / 128, MROWS / 128), 256, 0, stream>>>(
      ctx, Wd, bd, out);
  add_ln_kernel<<<MROWS, 256, 0, stream>>>(out, input, lw, lb);
}
